// Head_46961172415001
// MI455X (gfx1250) — matrix operand order repaired and checked by reading, not compiled, not run on hardware
//
#include <hip/hip_runtime.h>
#include <stdint.h>

// CDNA5 (gfx1250) wave32 WMMA attention head.  B=8, T=2048, E=1024, H=64.
// Pipeline:
//   K0  conv_kernel : emb fp32 -> bf16 (one-time, memory-bound)
//   K0b wpack_kernel: Wq/Wk/Wv fp32 -> bf16 pre-packed in WMMA B-fragment layout
//   K1  qkv_gemm    : q,k (bf16 [B*T,64]) and v transposed (bf16 [B,64,T]);
//                     inner loop = 6x b128 loads + 2x v_wmma_f32_16x16x32_bf16
//   K2  attn_kernel : causal flash attention, S^T = K*Q^T orientation so the
//                     softmax is per-lane (no LDS transpose, minimal shuffles)

typedef __attribute__((ext_vector_type(16))) __bf16 v16bf;
typedef __attribute__((ext_vector_type(8)))  float  v8f;

#define BATCH 8
#define SEQ   2048
#define EMB   1024
#define HEAD  64
#define BT    (BATCH * SEQ)

union FragAB { v16bf v; uint32_t u[8]; uint4 q4[2]; };

__device__ __forceinline__ uint16_t f2bf(float x) {           // RNE
  uint32_t u = __float_as_uint(x);
  return (uint16_t)((u + 0x7FFFu + ((u >> 16) & 1u)) >> 16);
}
__device__ __forceinline__ uint32_t pack2(float lo, float hi) {
  return (uint32_t)f2bf(lo) | ((uint32_t)f2bf(hi) << 16);
}
__device__ __forceinline__ uint32_t pack2t(float lo, float hi) {  // truncate (P in [0,1])
  return (__float_as_uint(hi) & 0xffff0000u) | (__float_as_uint(lo) >> 16);
}

// ---------------------------------------------------------------------------
// K0: emb fp32 -> bf16, 8 elements/thread.  grid = BT*EMB/2048 = 8192
// ---------------------------------------------------------------------------
__global__ void __launch_bounds__(256)
conv_kernel(const float* __restrict__ emb, uint16_t* __restrict__ embb) {
  const size_t i = ((size_t)blockIdx.x * 256 + threadIdx.x) * 8;
  const float4 f0 = *(const float4*)(emb + i);
  const float4 f1 = *(const float4*)(emb + i + 4);
  uint4 o;
  o.x = pack2(f0.x, f0.y); o.y = pack2(f0.z, f0.w);
  o.z = pack2(f1.x, f1.y); o.w = pack2(f1.z, f1.w);
  *(uint4*)(embb + i) = o;
}

// ---------------------------------------------------------------------------
// K0b: pack W into per-lane B-fragment layout.
// wpk[which][e0tile(32)][n0tile(4)][lane(32)][j(8)] dwords (bf16 pairs).
// B-layout (16-bit, 32x16): lane<16 -> N=lane, K={0..7,16..23}; lane>=16 -> K={8..15,24..31}.
// grid = (128, 3) [x = e0tile*4 + n0tile], block = 32.
// ---------------------------------------------------------------------------
__global__ void __launch_bounds__(32)
wpack_kernel(const float* __restrict__ Wq, const float* __restrict__ Wk,
             const float* __restrict__ Wv, uint32_t* __restrict__ wpk) {
  const int which = blockIdx.y;
  const float* W = (which == 0) ? Wq : (which == 1) ? Wk : Wv;
  const int lane = threadIdx.x & 31;
  const int half = lane >> 4, ln = lane & 15;
  const int e0 = (blockIdx.x >> 2) * 32;
  const int n  = (blockIdx.x & 3) * 16 + ln;
  uint32_t* dst = wpk + (size_t)which * 32768 + (size_t)blockIdx.x * 256 + lane * 8;
#pragma unroll
  for (int j = 0; j < 8; ++j) {
    const float* p = W + (size_t)(e0 + half * 8 + 2 * j + (j & 4) * 2) * HEAD + n;
    dst[j] = pack2(p[0], p[HEAD]);
  }
}

// ---------------------------------------------------------------------------
// K1: q/k/vT GEMM.  grid = (BT/32, 3), block = 128 (4 waves, one 16-col tile
// each, two 16-row tiles sharing the B fragment).
// A-layout (16-bit, 16x32): lane<16 -> M=lane, K={0..7,16..23};
//                           lane>=16 -> M=lane-16, K={8..15,24..31}.
// ---------------------------------------------------------------------------
__global__ void __launch_bounds__(128)
qkv_gemm_kernel(const uint16_t* __restrict__ embb, const uint32_t* __restrict__ wpk,
                const float* __restrict__ bq, const float* __restrict__ bk,
                const float* __restrict__ bv, uint16_t* __restrict__ qkv) {
  const int wave = threadIdx.x >> 5;
  const int lane = threadIdx.x & 31;
  const int half = lane >> 4, ln = lane & 15;
  const int which = blockIdx.y;
  const float* bia = (which == 0) ? bq : (which == 1) ? bk : bv;
  uint16_t* dst = qkv + (size_t)which * BT * HEAD;

  const int r0 = blockIdx.x * 32;
  const int n  = wave * 16 + ln;

  const uint32_t* wp = wpk + (size_t)which * 32768 + wave * 256 + lane * 8;
  const uint16_t* a0 = embb + (size_t)(r0 + ln) * EMB + half * 8;
  const uint16_t* a1 = a0 + 16 * EMB;

  v8f acc0 = {0.f,0.f,0.f,0.f,0.f,0.f,0.f,0.f};
  v8f acc1 = {0.f,0.f,0.f,0.f,0.f,0.f,0.f,0.f};

  for (int e0 = 0; e0 < EMB; e0 += 32) {
    FragAB fa0, fa1, fb;
    fa0.q4[0] = *(const uint4*)(a0 + e0);
    fa0.q4[1] = *(const uint4*)(a0 + e0 + 16);
    fa1.q4[0] = *(const uint4*)(a1 + e0);
    fa1.q4[1] = *(const uint4*)(a1 + e0 + 16);
    fb.q4[0]  = *(const uint4*)(wp);
    fb.q4[1]  = *(const uint4*)(wp + 4);
    wp += 1024;
    acc0 = __builtin_amdgcn_wmma_f32_16x16x32_bf16(false, fa0.v, false, fb.v,
                                                   (short)0, acc0, false, false);
    acc1 = __builtin_amdgcn_wmma_f32_16x16x32_bf16(false, fa1.v, false, fb.v,
                                                   (short)0, acc1, false, false);
  }
  const float bb = bia[n];
  acc0 = acc0 + bb;
  acc1 = acc1 + bb;

  if (which < 2) {              // row-major [B*T, 64]
#pragma unroll
    for (int j = 0; j < 8; ++j) {
      dst[(size_t)(r0 + j + half * 8) * HEAD + n]      = f2bf(acc0[j]);
      dst[(size_t)(r0 + 16 + j + half * 8) * HEAD + n] = f2bf(acc1[j]);
    }
  } else {                      // transposed [B, 64, T]
    const int rb = r0 >> 11;
    const int t0 = (r0 & (SEQ - 1)) + half * 8;
    uint16_t* vdst = dst + (size_t)rb * HEAD * SEQ + (size_t)n * SEQ;
    uint4 p0, p1;
    p0.x = pack2(acc0[0], acc0[1]); p0.y = pack2(acc0[2], acc0[3]);
    p0.z = pack2(acc0[4], acc0[5]); p0.w = pack2(acc0[6], acc0[7]);
    p1.x = pack2(acc1[0], acc1[1]); p1.y = pack2(acc1[2], acc1[3]);
    p1.z = pack2(acc1[4], acc1[5]); p1.w = pack2(acc1[6], acc1[7]);
    *(uint4*)(vdst + t0)      = p0;
    *(uint4*)(vdst + t0 + 16) = p1;
  }
}

// ---------------------------------------------------------------------------
// K2: causal flash attention, transposed orientation S^T = K * Q^T.
// One wave per 16-query tile; lane = query, vector elements = keys/features.
// grid = 256, block = 128 (4 waves).  No LDS.
// ---------------------------------------------------------------------------
__global__ void __launch_bounds__(128)
attn_kernel(const uint16_t* __restrict__ q, const uint16_t* __restrict__ k,
            const uint16_t* __restrict__ vT, float* __restrict__ out) {
  const int wave = threadIdx.x >> 5;
  const int lane = threadIdx.x & 31;
  const int half = lane >> 4, ln = lane & 15;

  const int tile = blockIdx.x * 4 + wave;     // [0, 1024)
  const int b  = tile >> 7;
  const int q0 = (tile & 127) * 16;

  const uint16_t* qb = q  + (size_t)b * SEQ * HEAD;
  const uint16_t* kb = k  + (size_t)b * SEQ * HEAD;
  const uint16_t* vb = vT + (size_t)b * HEAD * SEQ;

  // Q^T as B-matrix: N = query = lane, K = feature (two 8-bf16 runs, +16 apart)
  FragAB qf[2];
#pragma unroll
  for (int t = 0; t < 2; ++t) {
    const uint16_t* qrow = qb + (size_t)(q0 + ln) * HEAD + t * 32 + half * 8;
    qf[t].q4[0] = *(const uint4*)(qrow);
    qf[t].q4[1] = *(const uint4*)(qrow + 16);
  }

  v8f OT[4];   // O^T tiles: M = feature (j + 8*half within nt*16), N = query
#pragma unroll
  for (int nt = 0; nt < 4; ++nt)
    OT[nt] = (v8f){0.f,0.f,0.f,0.f,0.f,0.f,0.f,0.f};

  float m = -1e30f, l = 0.f;                  // per-lane (per-query) stats
  const float scale = 0.03125f;               // EMB^-0.5
  const int qr = q0 + ln;                     // this lane's query row
  const int qmax = q0 + 15;

  for (int s0 = 0; s0 <= qmax; s0 += 32) {
    __builtin_prefetch(kb + (size_t)(s0 + 32 + ln) * HEAD, 0, 1);
    __builtin_prefetch(vb + (size_t)ln * SEQ + s0 + 32, 0, 1);

    // two 16-key S^T subtiles: A = K-tile (M=key, K=feature), B = Q^T
    v8f st[2];
#pragma unroll
    for (int sidx = 0; sidx < 2; ++sidx) {
      v8f s = {0.f,0.f,0.f,0.f,0.f,0.f,0.f,0.f};
#pragma unroll
      for (int t = 0; t < 2; ++t) {
        FragAB kf;
        const uint16_t* krow =
            kb + (size_t)(s0 + sidx * 16 + ln) * HEAD + t * 32 + half * 8;
        kf.q4[0] = *(const uint4*)(krow);
        kf.q4[1] = *(const uint4*)(krow + 16);
        s = __builtin_amdgcn_wmma_f32_16x16x32_bf16(false, kf.v, false, qf[t].v,
                                                    (short)0, s, false, false);
      }
      st[sidx] = s;
    }

    // mask + scale: element (sidx, j) is key s0 + sidx*16 + j + 8*half
    float xe[16];
#pragma unroll
    for (int sidx = 0; sidx < 2; ++sidx)
#pragma unroll
      for (int j = 0; j < 8; ++j) {
        const int key = s0 + sidx * 16 + j + half * 8;
        xe[sidx * 8 + j] = (key <= qr) ? st[sidx][j] * scale : -1e30f;
      }

    // per-query max: in-lane over 16 elements, then combine lane halves
    float rm = xe[0];
#pragma unroll
    for (int e = 1; e < 16; ++e) rm = fmaxf(rm, xe[e]);
    rm = fmaxf(rm, __shfl_xor(rm, 16, 32));
    const float mnew = fmaxf(m, rm);
    const float corr = __expf(m - mnew);
    m = mnew;

    float pe[16], ps = 0.f;
#pragma unroll
    for (int e = 0; e < 16; ++e) { pe[e] = __expf(xe[e] - mnew); ps += pe[e]; }
    ps += __shfl_xor(ps, 16, 32);
    l = l * corr + ps;
#pragma unroll
    for (int nt = 0; nt < 4; ++nt) OT[nt] = OT[nt] * corr;

    // build P^T B-fragment (K = key-in-chunk, N = query) from registers:
    // own half holds keys {8*half..8*half+7} of each subtile; swap halves.
    uint32_t u0[4], u1[4], o0[4], o1[4];
#pragma unroll
    for (int j2 = 0; j2 < 4; ++j2) {
      u0[j2] = pack2t(pe[2 * j2],     pe[2 * j2 + 1]);
      u1[j2] = pack2t(pe[8 + 2 * j2], pe[9 + 2 * j2]);
      o0[j2] = (uint32_t)__shfl_xor((int)u0[j2], 16, 32);
      o1[j2] = (uint32_t)__shfl_xor((int)u1[j2], 16, 32);
    }
    FragAB pf;
#pragma unroll
    for (int j2 = 0; j2 < 4; ++j2) {
      pf.u[j2]     = half ? o1[j2] : u0[j2];   // keys 0..7   / 16..23 (lane half 0 / 1)
      pf.u[4 + j2] = half ? u1[j2] : o0[j2];   // keys 8..15  / 24..31 (lane half 0 / 1)
    }

    // O^T += V^T * P^T  (A = V^T tile: M=feature, K=keys, contiguous b128s)
#pragma unroll
    for (int nt = 0; nt < 4; ++nt) {
      FragAB vf;
      const uint16_t* vrow = vb + (size_t)(nt * 16 + ln) * SEQ + s0 + half * 16;
      vf.q4[0] = *(const uint4*)(vrow);
      vf.q4[1] = *(const uint4*)(vrow + 8);
      OT[nt] = __builtin_amdgcn_wmma_f32_16x16x32_bf16(false, vf.v, false, pf.v,
                                                       (short)0, OT[nt], false, false);
    }
  }

  // normalize and store: lane writes its query row, features j + 8*half + nt*16
  const float inv = 1.0f / l;
  float* orow = out + ((size_t)b * SEQ + qr) * HEAD;
#pragma unroll
  for (int nt = 0; nt < 4; ++nt)
#pragma unroll
    for (int j = 0; j < 8; ++j)
      orow[nt * 16 + j + half * 8] = OT[nt][j] * inv;
}

// ---------------------------------------------------------------------------
extern "C" void kernel_launch(void* const* d_in, const int* in_sizes, int n_in,
                              void* d_out, int out_size, void* d_ws, size_t ws_size,
                              hipStream_t stream) {
  (void)in_sizes; (void)n_in; (void)out_size; (void)ws_size;
  const float* emb = (const float*)d_in[0];
  const float* Wq  = (const float*)d_in[1];
  const float* bq  = (const float*)d_in[2];
  const float* Wk  = (const float*)d_in[3];
  const float* bk  = (const float*)d_in[4];
  const float* Wv  = (const float*)d_in[5];
  const float* bv  = (const float*)d_in[6];
  float* out = (float*)d_out;

  // workspace: embb(32MB) | q(2MB) | k(2MB) | vT(2MB) | wpack(384KB)
  uint16_t* embb = (uint16_t*)d_ws;
  uint16_t* qw   = embb + (size_t)BT * EMB;
  uint16_t* kw   = qw + (size_t)BT * HEAD;
  uint16_t* vw   = kw + (size_t)BT * HEAD;
  uint32_t* wpk  = (uint32_t*)(vw + (size_t)BT * HEAD);

  conv_kernel<<<dim3((BT * EMB) / 2048), dim3(256), 0, stream>>>(emb, embb);
  wpack_kernel<<<dim3(128, 3), dim3(32), 0, stream>>>(Wq, Wk, Wv, wpk);
  qkv_gemm_kernel<<<dim3(BT / 32, 3), dim3(128), 0, stream>>>(embb, wpk, bq, bk, bv, qw);
  attn_kernel<<<dim3(256), dim3(128), 0, stream>>>(qw, kw, vw, out);
}
